// GraphSAGE_14525579395820
// MI455X (gfx1250) — compile-verified, matrix operand order checked by reading, not run on hardware
//
#include <hip/hip_runtime.h>
#include <hip/hip_bf16.h>
#include <math.h>

// ---------------- problem constants (match reference) ----------------
#define NN   10000
#define TT   32
#define DD   128
#define HH   128
#define EE   320000
#define EE2  160000
#define SLOPE 0.01f

typedef __attribute__((ext_vector_type(16))) _Float16 v16h;
typedef __attribute__((ext_vector_type(8)))  float    v8f;

// =====================================================================
// B pre-pack: convert f32 weight matrix into f16 WMMA B-fragment layout.
//   out[((ntile*kchunks + kc)*32 + lane)*16 + e]
//   lane = half*16 + r : column n = ntile*16 + r
//   e<8  -> k = kc*32 + half*8 + e
//   e>=8 -> k = kc*32 + half*8 + 8 + e      (i.e. +16 + (e-8))
//   bt==1: B stored [N,K] (torch W for x@W.T), else [K,N]
// Runs once per weight per launch (negligible).
// =====================================================================
__global__ __launch_bounds__(32)
void pack_b_kernel(const float* __restrict__ B, int ldb, int bt, int K, int Ntot,
                   _Float16* __restrict__ out)
{
    const int lane = threadIdx.x;
    const int half = lane >> 4, r = lane & 15;
    const int nt = blockIdx.x, kc = blockIdx.y;
    int n = nt * 16 + r; if (n >= Ntot) n = Ntot - 1;
    const int kb = kc * 32 + half * 8;
    _Float16* dst = out + (((size_t)nt * gridDim.y + kc) * 32 + lane) * 16;
#pragma unroll
    for (int e = 0; e < 16; ++e) {
        int k = kb + (e < 8 ? e : 8 + e);
        float v = (k < K) ? (bt ? B[(size_t)n * ldb + k] : B[(size_t)k * ldb + n]) : 0.0f;
        dst[e] = (_Float16)v;
    }
}

// =====================================================================
// Wave-level WMMA GEMM with packed B:
//   C[M,N] = act( [C +] A_gathered[M,K] @ B + bias )
// Block = 128 threads = 4 waves; wave w computes a 16x64 strip:
//   rows  m0 = (blockIdx.y*4 + w)*16 .. +15
//   cols  n0 = blockIdx.x*64 .. +63   (four 16-wide B fragments, A reused 4x)
// K consumed 32/chunk via v_wmma_f32_16x16x32_f16 (f32 accumulate).
// Requires: K % 32 == 0, Ntot % 64 == 0 (true for all calls here).
// =====================================================================
__global__ __launch_bounds__(128)
void wmma_gemm_kernel(const float* __restrict__ A,
                      const int* __restrict__ aidx, int aidx_stride, int aidx_off,
                      int lda, int aoff,
                      const _Float16* __restrict__ Bp, int kchunks,
                      float* __restrict__ C, int ldc,
                      const float* __restrict__ bias,
                      int M, int Ntot, int acc, int act)
{
    const int lane = threadIdx.x & 31;
    const int wave = threadIdx.x >> 5;
    const int half = lane >> 4, r = lane & 15;
    const int mtile = blockIdx.y * 4 + wave;
    const int m0 = mtile * 16;
    if (m0 >= M) return;                       // uniform per wave
    const int n0 = blockIdx.x * 64;

    // A row pointer (clamped; M is a multiple of 16 in all calls)
    int m_a = m0 + r; if (m_a >= M) m_a = M - 1;
    const int row = aidx ? aidx[(size_t)m_a * aidx_stride + aidx_off] : m_a;
    const float* Arow = A + (size_t)row * lda + aoff + half * 8;

    // packed B fragments for the four column tiles
    const v16h* B0 = (const v16h*)Bp + ((size_t)(4 * blockIdx.x + 0) * kchunks) * 32 + lane;
    const v16h* B1 = (const v16h*)Bp + ((size_t)(4 * blockIdx.x + 1) * kchunks) * 32 + lane;
    const v16h* B2 = (const v16h*)Bp + ((size_t)(4 * blockIdx.x + 2) * kchunks) * 32 + lane;
    const v16h* B3 = (const v16h*)Bp + ((size_t)(4 * blockIdx.x + 3) * kchunks) * 32 + lane;

    v8f c0 = {}, c1 = {}, c2 = {}, c3 = {};
#pragma unroll 2
    for (int kc = 0; kc < kchunks; ++kc) {
        const float4* pa = (const float4*)(Arow + kc * 32);
        float4 x0 = pa[0], x1 = pa[1], x2 = pa[4], x3 = pa[5];
        v16h av;
        av[0]=(_Float16)x0.x; av[1]=(_Float16)x0.y; av[2]=(_Float16)x0.z; av[3]=(_Float16)x0.w;
        av[4]=(_Float16)x1.x; av[5]=(_Float16)x1.y; av[6]=(_Float16)x1.z; av[7]=(_Float16)x1.w;
        av[8]=(_Float16)x2.x; av[9]=(_Float16)x2.y; av[10]=(_Float16)x2.z; av[11]=(_Float16)x2.w;
        av[12]=(_Float16)x3.x; av[13]=(_Float16)x3.y; av[14]=(_Float16)x3.z; av[15]=(_Float16)x3.w;
        v16h b0 = B0[(size_t)kc * 32];
        v16h b1 = B1[(size_t)kc * 32];
        v16h b2 = B2[(size_t)kc * 32];
        v16h b3 = B3[(size_t)kc * 32];
        c0 = __builtin_amdgcn_wmma_f32_16x16x32_f16(false, av, false, b0, (short)0, c0, false, false);
        c1 = __builtin_amdgcn_wmma_f32_16x16x32_f16(false, av, false, b1, (short)0, c1, false, false);
        c2 = __builtin_amdgcn_wmma_f32_16x16x32_f16(false, av, false, b2, (short)0, c2, false, false);
        c3 = __builtin_amdgcn_wmma_f32_16x16x32_f16(false, av, false, b3, (short)0, c3, false, false);
    }

    // Epilogue: lane r holds columns n0 + 16*f + r, rows m0 + v + half*8
#pragma unroll
    for (int v = 0; v < 8; ++v) {
        const int m = m0 + v + half * 8;
        if (m < M) {
            float* crow = C + (size_t)m * ldc;
            float vv[4] = { c0[v], c1[v], c2[v], c3[v] };
#pragma unroll
            for (int f = 0; f < 4; ++f) {
                const int n = n0 + 16 * f + r;
                float val = vv[f];
                if (acc)  val += crow[n];
                if (bias) val += bias[n];
                if (act)  val = (val > 0.0f) ? val : SLOPE * val;
                crow[n] = val;
            }
        }
    }
}

// ---------------- elementwise / scatter helpers ----------------
__global__ void fill_kernel(float* __restrict__ p, float v, size_t n) {
    size_t i = (size_t)blockIdx.x * blockDim.x + threadIdx.x;
    if (i < n) p[i] = v;
}

__global__ void lstm_cell_kernel(const float* __restrict__ g, const float* __restrict__ b,
                                 const int* __restrict__ mask, int t,
                                 float* __restrict__ h, float* __restrict__ c, int n_total)
{
    int i = blockIdx.x * blockDim.x + threadIdx.x;
    if (i >= n_total) return;
    int n = i / HH, j = i % HH;
    const float* gr = g + (size_t)n * 4 * HH;
    float gi = gr[j]          + b[j];
    float gf = gr[HH + j]     + b[HH + j];
    float gg = gr[2*HH + j]   + b[2*HH + j];
    float go = gr[3*HH + j]   + b[3*HH + j];
    float si = 1.0f / (1.0f + __expf(-gi));
    float sf = 1.0f / (1.0f + __expf(-gf));
    float so = 1.0f / (1.0f + __expf(-go));
    float cn = sf * c[i] + si * tanhf(gg);
    float hn = so * tanhf(cn);
    float mt = (float)mask[(size_t)n * TT + t];
    h[i] = mt * hn + (1.0f - mt) * h[i];
    c[i] = mt * cn + (1.0f - mt) * c[i];
}

__global__ void concat2_kernel(const float* __restrict__ a, const float* __restrict__ b,
                               float* __restrict__ out, int n_total)
{
    int i = blockIdx.x * blockDim.x + threadIdx.x;
    if (i >= n_total) return;
    int n = i / (2*DD), j = i % (2*DD);
    out[i] = (j < DD) ? a[(size_t)n*DD + j] : b[(size_t)n*DD + (j - DD)];
}

__global__ void gather_rows_kernel(const float* __restrict__ emb, const int* __restrict__ idx,
                                   float* __restrict__ out, int n_total)
{
    int i = blockIdx.x * blockDim.x + threadIdx.x;
    if (i >= n_total) return;
    int n = i / DD, d = i % DD;
    out[i] = emb[(size_t)idx[n] * DD + d];
}

__global__ void count_deg_kernel(const int* __restrict__ dst, float* __restrict__ deg, int e_total)
{
    int e = blockIdx.x * blockDim.x + threadIdx.x;
    if (e >= e_total) return;
    atomicAdd(&deg[dst[e]], 1.0f);
}

__global__ void scatter_add_kernel(const float* __restrict__ h,
                                   const int* __restrict__ src, const int* __restrict__ dst,
                                   float* __restrict__ msum, int e_total, int fdim)
{
    size_t i = (size_t)blockIdx.x * blockDim.x + threadIdx.x;
    if (i >= (size_t)e_total * fdim) return;
    int e = (int)(i / fdim), f = (int)(i % fdim);
    atomicAdd(&msum[(size_t)dst[e] * fdim + f], h[(size_t)src[e] * fdim + f]);
}

__global__ void div_deg_kernel(float* __restrict__ msum, const float* __restrict__ deg,
                               int n_total, int fdim)
{
    size_t i = (size_t)blockIdx.x * blockDim.x + threadIdx.x;
    if (i >= (size_t)n_total * fdim) return;
    int n = (int)(i / fdim);
    msum[i] = msum[i] / fmaxf(deg[n], 1.0f);
}

__global__ void attn_logits_kernel(const float* __restrict__ z, const float* __restrict__ attn_W,
                                   const float* __restrict__ attn_b, float* __restrict__ logits)
{
    int i = blockIdx.x * blockDim.x + threadIdx.x;
    if (i >= NN * 5) return;
    int n = i / 5, s = i % 5;
    const float* zr = z + ((size_t)s * NN + n) * HH;
    float acc = attn_b[0];
    for (int h = 0; h < HH; ++h) acc += zr[h] * attn_W[h];
    logits[i] = acc;
}

__global__ void softmax5_kernel(const float* __restrict__ logits, float* __restrict__ a)
{
    int n = blockIdx.x * blockDim.x + threadIdx.x;
    if (n >= NN) return;
    float v[5], mx = -1e30f;
    for (int s = 0; s < 5; ++s) { v[s] = logits[n*5+s]; mx = fmaxf(mx, v[s]); }
    float den = 0.0f;
    for (int s = 0; s < 5; ++s) { v[s] = __expf(v[s] - mx); den += v[s]; }
    for (int s = 0; s < 5; ++s) a[n*5+s] = v[s] / den;
}

__global__ void fuse_kernel(const float* __restrict__ z, const float* __restrict__ a,
                            float* __restrict__ hf)
{
    size_t i = (size_t)blockIdx.x * blockDim.x + threadIdx.x;
    if (i >= (size_t)NN * HH) return;
    int n = (int)(i / HH), f = (int)(i % HH);
    float acc = 0.0f;
    for (int s = 0; s < 5; ++s)
        acc += z[((size_t)s * NN + n) * HH + f] * a[n*5+s];
    hf[i] = acc;
}

__global__ void score_kernel(const float* __restrict__ t1, const float* __restrict__ fco_W,
                             const float* __restrict__ fco_b, float* __restrict__ out)
{
    size_t i = (size_t)blockIdx.x * blockDim.x + threadIdx.x;
    if (i >= (size_t)EE2 * 2) return;
    int e = (int)(i / 2), o = (int)(i % 2);
    const float* tr = t1 + (size_t)e * HH;
    float acc = fco_b[o];
    for (int h = 0; h < HH; ++h) acc += tr[h] * fco_W[h*2 + o];
    out[i] = acc;
}

__global__ void attn_edge_kernel(const float* __restrict__ a, const int* __restrict__ e_src,
                                 float* __restrict__ out)
{
    size_t i = (size_t)blockIdx.x * blockDim.x + threadIdx.x;
    if (i >= (size_t)EE2 * 10) return;
    int e = (int)(i / 10), s = (int)(i % 5);
    out[i] = a[(size_t)e_src[e] * 5 + s];
}

// ---------------- host-side helpers ----------------
static inline const _Float16* pack_b(hipStream_t st, _Float16* base, size_t& hoff,
                                     const float* B, int ldb, int bt, int K, int Ntot)
{
    const int nt = (Ntot + 15) / 16;
    const int kc = (K + 31) / 32;
    hoff = (hoff + 127) & ~(size_t)127;            // 256B alignment
    _Float16* dst = base + hoff;
    hoff += (size_t)nt * kc * 512;                 // 32 lanes * 16 halves per tile-chunk
    dim3 grid(nt, kc);
    pack_b_kernel<<<grid, 32, 0, st>>>(B, ldb, bt, K, Ntot, dst);
    return dst;
}

static inline void launch_gemm(hipStream_t st,
                               const float* A, const int* aidx, int aidx_stride, int aidx_off,
                               int lda, int aoff,
                               const _Float16* Bp,
                               float* C, int ldc, const float* bias,
                               int M, int Ntot, int K, int acc, int act)
{
    const int kchunks = (K + 31) / 32;
    dim3 grid(Ntot / 64, (M / 16 + 3) / 4);
    wmma_gemm_kernel<<<grid, 128, 0, st>>>(A, aidx, aidx_stride, aidx_off, lda, aoff,
                                           Bp, kchunks, C, ldc, bias, M, Ntot, acc, act);
}

static inline void launch_fill(hipStream_t st, float* p, float v, size_t n)
{
    fill_kernel<<<(unsigned)((n + 255) / 256), 256, 0, st>>>(p, v, n);
}

extern "C" void kernel_launch(void* const* d_in, const int* in_sizes, int n_in,
                              void* d_out, int out_size, void* d_ws, size_t ws_size,
                              hipStream_t stream)
{
    // ---- inputs (setup_inputs order) ----
    const int*   inputs_s  = (const int*)  d_in[0];
    const int*   inputs_sm = (const int*)  d_in[1];
    const int*   inputs_c  = (const int*)  d_in[2];
    const int*   inputs_co = (const int*)  d_in[3];
    const int*   inputs_sl = (const int*)  d_in[4];
    const float* inputs_ip = (const float*)d_in[5];
    const int*   blk_src   = (const int*)  d_in[6];
    const int*   blk_dst   = (const int*)  d_in[7];
    const int*   e_src     = (const int*)  d_in[8];
    const int*   e_dst     = (const int*)  d_in[9];
    const float* emb_url   = (const float*)d_in[10];
    const float* emb_cat   = (const float*)d_in[11];
    const float* emb_cou   = (const float*)d_in[12];
    const float* emb_sl_t  = (const float*)d_in[13];
    const float* Wih_f     = (const float*)d_in[14];
    const float* Whh_f     = (const float*)d_in[15];
    const float* b_f       = (const float*)d_in[16];
    const float* Wih_b     = (const float*)d_in[17];
    const float* Whh_b     = (const float*)d_in[18];
    const float* b_b       = (const float*)d_in[19];
    const float* fc_lstm_W = (const float*)d_in[20];
    const float* fc_lstm_b = (const float*)d_in[21];
    const float* attnlin_W = (const float*)d_in[22];
    const float* attn_W    = (const float*)d_in[23];
    const float* attn_b    = (const float*)d_in[24];
    const float* W0_self   = (const float*)d_in[25];
    const float* W0_neigh  = (const float*)d_in[26];
    const float* b0        = (const float*)d_in[27];
    const float* Wip_self  = (const float*)d_in[28];
    const float* Wip_neigh = (const float*)d_in[29];
    const float* bip       = (const float*)d_in[30];
    const float* W1_self   = (const float*)d_in[31];
    const float* W1_neigh  = (const float*)d_in[32];
    const float* b1        = (const float*)d_in[33];
    const float* fc_W      = (const float*)d_in[34];
    const float* fc_b      = (const float*)d_in[35];
    const float* fco_W     = (const float*)d_in[36];
    const float* fco_b     = (const float*)d_in[37];
    (void)in_sizes; (void)n_in; (void)out_size; (void)ws_size;

    // ---- workspace: float region ----
    float* ws = (float*)d_ws;
    size_t off = 0;
    float* h_f   = ws + off; off += (size_t)NN * HH;
    float* c_f   = ws + off; off += (size_t)NN * HH;
    float* h_b   = ws + off; off += (size_t)NN * HH;
    float* c_b   = ws + off; off += (size_t)NN * HH;
    float* g     = ws + off; off += (size_t)NN * 4 * HH;
    float* catb  = ws + off; off += (size_t)NN * 2 * DD;
    float* F0    = ws + off; off += (size_t)4 * NN * DD;
    float* msum  = ws + off; off += (size_t)NN * HH;
    float* deg   = ws + off; off += (size_t)NN + 64;
    float* S0    = ws + off; off += (size_t)5 * NN * HH;
    float* S1    = ws + off; off += (size_t)5 * NN * HH;
    float* z     = ws + off; off += (size_t)5 * NN * HH;
    float* logits= ws + off; off += (size_t)NN * 5 + 64;
    float* a     = ws + off; off += (size_t)NN * 5 + 64;
    float* hfused= ws + off; off += (size_t)NN * HH;
    float* t1    = ws + off; off += (size_t)EE2 * HH;
    off = (off + 63) & ~(size_t)63;

    // ---- workspace: packed f16 weight region ----
    _Float16* hbase = (_Float16*)(ws + off);
    size_t hoff = 0;
    const _Float16* pWih_f = pack_b(stream, hbase, hoff, Wih_f, DD, 1, DD, 4*HH);
    const _Float16* pWhh_f = pack_b(stream, hbase, hoff, Whh_f, HH, 1, HH, 4*HH);
    const _Float16* pWih_b = pack_b(stream, hbase, hoff, Wih_b, DD, 1, DD, 4*HH);
    const _Float16* pWhh_b = pack_b(stream, hbase, hoff, Whh_b, HH, 1, HH, 4*HH);
    const _Float16* pFcL   = pack_b(stream, hbase, hoff, fc_lstm_W, DD, 0, 2*DD, DD);
    const _Float16* pW0s[4], *pW0n[4];
    for (int s = 0; s < 4; ++s) {
        pW0s[s] = pack_b(stream, hbase, hoff, W0_self  + (size_t)s*DD*HH, HH, 0, DD, HH);
        pW0n[s] = pack_b(stream, hbase, hoff, W0_neigh + (size_t)s*DD*HH, HH, 0, DD, HH);
    }
    const _Float16* pWipS = pack_b(stream, hbase, hoff, Wip_self,  HH, 0, 32, HH);
    const _Float16* pWipN = pack_b(stream, hbase, hoff, Wip_neigh, HH, 0, 32, HH);
    const _Float16* pW1s[5], *pW1n[5];
    for (int s = 0; s < 5; ++s) {
        pW1s[s] = pack_b(stream, hbase, hoff, W1_self  + (size_t)s*HH*HH, HH, 0, HH, HH);
        pW1n[s] = pack_b(stream, hbase, hoff, W1_neigh + (size_t)s*HH*HH, HH, 0, HH, HH);
    }
    const _Float16* pAttn = pack_b(stream, hbase, hoff, attnlin_W, HH, 0, HH, HH);
    const _Float16* pFcA  = pack_b(stream, hbase, hoff, fc_W,                 HH, 0, HH, HH);
    const _Float16* pFcB  = pack_b(stream, hbase, hoff, fc_W + (size_t)HH*HH, HH, 0, HH, HH);

    float* out = (float*)d_out;
    const int NH = NN * HH;

    // ---------------- bi-LSTM over URL tokens ----------------
    launch_fill(stream, h_f, 0.0f, (size_t)4 * NH);   // h_f,c_f,h_b,c_b contiguous

    for (int t = 0; t < TT; ++t) {                    // forward
        launch_gemm(stream, emb_url, inputs_s, TT, t, DD, 0,
                    pWih_f, g, 4*HH, nullptr, NN, 4*HH, DD, 0, 0);
        launch_gemm(stream, h_f, nullptr, 0, 0, HH, 0,
                    pWhh_f, g, 4*HH, nullptr, NN, 4*HH, HH, 1, 0);
        lstm_cell_kernel<<<(NH + 255)/256, 256, 0, stream>>>(g, b_f, inputs_sm, t, h_f, c_f, NH);
    }
    for (int t = TT - 1; t >= 0; --t) {               // backward
        launch_gemm(stream, emb_url, inputs_s, TT, t, DD, 0,
                    pWih_b, g, 4*HH, nullptr, NN, 4*HH, DD, 0, 0);
        launch_gemm(stream, h_b, nullptr, 0, 0, HH, 0,
                    pWhh_b, g, 4*HH, nullptr, NN, 4*HH, HH, 1, 0);
        lstm_cell_kernel<<<(NH + 255)/256, 256, 0, stream>>>(g, b_b, inputs_sm, t, h_b, c_b, NH);
    }

    // vec_url = lrelu(concat(h_f,h_b) @ fc_lstm_W + fc_lstm_b) -> F0 stream 0
    concat2_kernel<<<(NN*2*DD + 255)/256, 256, 0, stream>>>(h_f, h_b, catb, NN*2*DD);
    launch_gemm(stream, catb, nullptr, 0, 0, 2*DD, 0,
                pFcL, F0, DD, fc_lstm_b, NN, DD, 2*DD, 0, 1);

    gather_rows_kernel<<<(NN*DD + 255)/256, 256, 0, stream>>>(emb_cat,  inputs_c,  F0 + 1*(size_t)NN*DD, NN*DD);
    gather_rows_kernel<<<(NN*DD + 255)/256, 256, 0, stream>>>(emb_cou,  inputs_co, F0 + 2*(size_t)NN*DD, NN*DD);
    gather_rows_kernel<<<(NN*DD + 255)/256, 256, 0, stream>>>(emb_sl_t, inputs_sl, F0 + 3*(size_t)NN*DD, NN*DD);

    launch_fill(stream, deg, 0.0f, NN);
    count_deg_kernel<<<(EE + 255)/256, 256, 0, stream>>>(blk_dst, deg, EE);

    // ---------------- SAGE layer 0 ----------------
    for (int s = 0; s < 5; ++s) {
        const float*    hin  = (s < 4) ? (F0 + (size_t)s * NN * DD) : inputs_ip;
        const int       fdim = (s < 4) ? DD : 32;
        const _Float16* Wsf  = (s < 4) ? pW0s[s] : pWipS;
        const _Float16* Wng  = (s < 4) ? pW0n[s] : pWipN;
        const float*    bb   = (s < 4) ? (b0 + (size_t)s * HH) : bip;
        float* hout = S0 + (size_t)s * NN * HH;

        launch_fill(stream, msum, 0.0f, (size_t)NN * fdim);
        scatter_add_kernel<<<(unsigned)(((size_t)EE*fdim + 255)/256), 256, 0, stream>>>(
            hin, blk_src, blk_dst, msum, EE, fdim);
        div_deg_kernel<<<(unsigned)(((size_t)NN*fdim + 255)/256), 256, 0, stream>>>(
            msum, deg, NN, fdim);
        launch_gemm(stream, hin,  nullptr, 0, 0, fdim, 0, Wsf, hout, HH, nullptr,
                    NN, HH, fdim, 0, 0);
        launch_gemm(stream, msum, nullptr, 0, 0, fdim, 0, Wng, hout, HH, bb,
                    NN, HH, fdim, 1, 1);
    }

    // ---------------- SAGE layer 1 ----------------
    for (int s = 0; s < 5; ++s) {
        const float* hin = S0 + (size_t)s * NN * HH;
        float* hout = S1 + (size_t)s * NN * HH;
        launch_fill(stream, msum, 0.0f, (size_t)NN * HH);
        scatter_add_kernel<<<(unsigned)(((size_t)EE*HH + 255)/256), 256, 0, stream>>>(
            hin, blk_src, blk_dst, msum, EE, HH);
        div_deg_kernel<<<(unsigned)(((size_t)NN*HH + 255)/256), 256, 0, stream>>>(
            msum, deg, NN, HH);
        launch_gemm(stream, hin,  nullptr, 0, 0, HH, 0, pW1s[s], hout, HH, nullptr,
                    NN, HH, HH, 0, 0);
        launch_gemm(stream, msum, nullptr, 0, 0, HH, 0, pW1n[s], hout, HH, b1 + (size_t)s*HH,
                    NN, HH, HH, 1, 1);
    }

    // ---------------- attention fusion ----------------
    for (int s = 0; s < 5; ++s)
        launch_gemm(stream, S1 + (size_t)s * NN * HH, nullptr, 0, 0, HH, 0,
                    pAttn, z + (size_t)s * NN * HH, HH, nullptr, NN, HH, HH, 0, 0);
    attn_logits_kernel<<<(NN*5 + 255)/256, 256, 0, stream>>>(z, attn_W, attn_b, logits);
    softmax5_kernel<<<(NN + 255)/256, 256, 0, stream>>>(logits, a);
    fuse_kernel<<<(NH + 255)/256, 256, 0, stream>>>(z, a, hfused);

    // ---------------- per-edge gather-concat MLP ----------------
    launch_gemm(stream, hfused, e_src, 1, 0, HH, 0,
                pFcA, t1, HH, nullptr, EE2, HH, HH, 0, 0);
    launch_gemm(stream, hfused, e_dst, 1, 0, HH, 0,
                pFcB, t1, HH, fc_b, EE2, HH, HH, 1, 1);

    score_kernel<<<(unsigned)(((size_t)EE2*2 + 255)/256), 256, 0, stream>>>(t1, fco_W, fco_b, out);
    attn_edge_kernel<<<(unsigned)(((size_t)EE2*10 + 255)/256), 256, 0, stream>>>(
        a, e_src, out + (size_t)2*EE2);
}
